// CRF_40699110097602
// MI455X (gfx1250) — matrix operand order repaired and checked by reading, not compiled, not run on hardware
//
#include <hip/hip_runtime.h>
#include <hip/hip_fp16.h>

typedef __attribute__((ext_vector_type(16))) _Float16 v16h;
typedef __attribute__((ext_vector_type(8)))  float    v8f;

#define NTAGS   64
#define SEQ     512
#define BATCH   512
#define TILE_B  16
#define NWAVE   4
#define THREADS (NWAVE * 32)
#define TDIM    66   // transitions is (NTAGS+2) x (NTAGS+2)

union Frag16 { v16h v; _Float16 h[16]; };

// ---------------------------------------------------------------------------
// Kernel 1: gold path score, one wave32 per batch row.
// ---------------------------------------------------------------------------
__global__ __launch_bounds__(256) void crf_gold_kernel(
    const float* __restrict__ em, const int* __restrict__ tags,
    const float* __restrict__ mask, const float* __restrict__ tr,
    float* __restrict__ gold) {
  const int wave = threadIdx.x >> 5;
  const int lane = threadIdx.x & 31;
  const int b = blockIdx.x * (blockDim.x >> 5) + wave;
  if (b >= BATCH) return;
  const int*   tg = tags + b * SEQ;
  const float* mk = mask + b * SEQ;
  const float* eb = em + (size_t)b * SEQ * NTAGS;

  float acc = 0.f, lensum = 0.f;
  for (int s = lane; s < SEQ; s += 32) {
    float m = mk[s];
    lensum += m;
    if (s == 0) {
      int first = tg[0];
      acc += (eb[first] + tr[first * TDIM + NTAGS]) * m;   // transitions[first, start]
    } else {
      int curr = tg[s], prev = tg[s - 1];
      acc += (tr[curr * TDIM + prev] + eb[s * NTAGS + curr]) * m;
    }
  }
  for (int off = 16; off; off >>= 1) {
    acc    += __shfl_down(acc, off, 32);
    lensum += __shfl_down(lensum, off, 32);
  }
  if (lane == 0) {
    int len  = (int)(lensum + 0.5f);
    int last = tg[len - 1];
    gold[b] = acc + tr[(NTAGS + 1) * TDIM + last];          // transitions[stop, last]
  }
}

// ---------------------------------------------------------------------------
// Kernel 2: forward scan via logsumexp-as-matmul on v_wmma_f32_16x16x32_f16.
// One WG per 16-row batch tile; wave w owns output columns [16w, 16w+16).
// ---------------------------------------------------------------------------
__global__ __launch_bounds__(THREADS) void crf_scan_kernel(
    const float* __restrict__ em, const float* __restrict__ mask,
    const float* __restrict__ tr, const float* __restrict__ gold,
    float* __restrict__ partial) {
  __shared__ float    sc[TILE_B][NTAGS];       // running scores (f32)
  __shared__ float    emitbuf[TILE_B][NTAGS];  // this step's emissions
  __shared__ _Float16 pA[TILE_B][NTAGS];       // exp(score - rowmax) as f16
  __shared__ float    rowmax[TILE_B];
  __shared__ float    maskbuf[TILE_B];
  __shared__ float    lz[TILE_B];

  const int tid  = threadIdx.x;
  const int wave = tid >> 5;
  const int lane = tid & 31;
  const int half = lane >> 4;     // 0: lanes 0-15, 1: lanes 16-31
  const int l    = lane & 15;
  const int b0   = blockIdx.x * TILE_B;

  // B fragments: E[k][n] = exp(transitions[k][n]); constant over the scan.
  // Layout (32x16 f16 B): e -> k = 32g + (e<8?0:16) + 8*half + (e&7), matching A.
  Frag16 bf[2];
#pragma unroll
  for (int g = 0; g < 2; ++g)
#pragma unroll
    for (int e = 0; e < 16; ++e) {
      int k = 32 * g + ((e < 8) ? 0 : 16) + 8 * half + (e & 7);
      int n = 16 * wave + l;
      bf[g].h[e] = (_Float16)__expf(tr[k * TDIM + n]);
    }

  // score0 = (emissions[:,0] + transitions[:T, start]) * mask[:,0]
  for (int i = tid; i < TILE_B * NTAGS; i += THREADS) {
    int r = i >> 6, c = i & 63;
    float m0 = mask[(b0 + r) * SEQ];
    sc[r][c] = (em[((size_t)(b0 + r) * SEQ) * NTAGS + c] + tr[c * TDIM + NTAGS]) * m0;
  }
  __syncthreads();

  for (int s = 1; s < SEQ; ++s) {
    // ---- phase 1: load emissions/mask; rowmax + exp(score - max) -> pA ----
    for (int i = tid; i < TILE_B * NTAGS; i += THREADS) {
      int r = i >> 6, c = i & 63;
      emitbuf[r][c] = em[((size_t)(b0 + r) * SEQ + s) * NTAGS + c];
    }
    if (tid < TILE_B) maskbuf[tid] = mask[(b0 + tid) * SEQ + s];
    if (s + 1 < SEQ && tid < 32) {   // prefetch next step (global_prefetch_b8)
      int r = tid >> 1, c = (tid & 1) * 32;
      __builtin_prefetch(&em[((size_t)(b0 + r) * SEQ + (s + 1)) * NTAGS + c], 0, 1);
    }
    {
      int r  = tid >> 3;           // 8 threads per row
      int c0 = (tid & 7) * 8;
      float v[8];
      float mx = -3.0e38f;
#pragma unroll
      for (int j = 0; j < 8; ++j) { v[j] = sc[r][c0 + j]; mx = fmaxf(mx, v[j]); }
#pragma unroll
      for (int off = 1; off < 8; off <<= 1) mx = fmaxf(mx, __shfl_xor(mx, off, 32));
      if ((tid & 7) == 0) rowmax[r] = mx;
#pragma unroll
      for (int j = 0; j < 8; ++j) pA[r][c0 + j] = (_Float16)__expf(v[j] - mx);
    }
    __syncthreads();

    // ---- phase 2: (16x64 P) @ (64x16 E-block) with two K=32 WMMAs ----
    // A 16x32 f16 layout: lane(l,half): e -> k = 32g + (e<8?0:16) + 2*((e>>1)&3) + (e&1) + 8*half
    Frag16 af[2];
#pragma unroll
    for (int g = 0; g < 2; ++g)
#pragma unroll
      for (int e = 0; e < 16; ++e) {
        int k = 32 * g + ((e < 8) ? 0 : 16) + ((e >> 1) & 3) * 2 + (e & 1) + 8 * half;
        af[g].h[e] = pA[l][k];
      }
    v8f c = {};
    c = __builtin_amdgcn_wmma_f32_16x16x32_f16(false, af[0].v, false, bf[0].v,
                                               (short)0, c, false, false);
    c = __builtin_amdgcn_wmma_f32_16x16x32_f16(false, af[1].v, false, bf[1].v,
                                               (short)0, c, false, false);

    // ---- phase 3: score = log(P@E) + rowmax + emit*mask ----
#pragma unroll
    for (int vv = 0; vv < 8; ++vv) {
      int r = vv + 8 * half;          // C layout: row = vgpr + 8*half
      int n = 16 * wave + l;          // col = lane & 15 within wave's block
      sc[r][n] = __logf(c[vv]) + rowmax[r] + emitbuf[r][n] * maskbuf[r];
    }
    __syncthreads();
  }

  // ---- finalize: logZ = LSE(score + transitions[stop, :T]) per row ----
  {
    int r  = tid >> 3;
    int c0 = (tid & 7) * 8;
    float v[8];
    float mx = -3.0e38f;
#pragma unroll
    for (int j = 0; j < 8; ++j) {
      v[j] = sc[r][c0 + j] + tr[(NTAGS + 1) * TDIM + c0 + j];
      mx = fmaxf(mx, v[j]);
    }
#pragma unroll
    for (int off = 1; off < 8; off <<= 1) mx = fmaxf(mx, __shfl_xor(mx, off, 32));
    float se = 0.f;
#pragma unroll
    for (int j = 0; j < 8; ++j) se += __expf(v[j] - mx);
#pragma unroll
    for (int off = 1; off < 8; off <<= 1) se += __shfl_xor(se, off, 32);
    if ((tid & 7) == 0) lz[r] = __logf(se) + mx;
  }
  __syncthreads();
  if (tid == 0) {
    float acc = 0.f;
    for (int r = 0; r < TILE_B; ++r) acc += lz[r] - gold[b0 + r];
    partial[blockIdx.x] = acc;   // deterministic: no float atomics
  }
}

// ---------------------------------------------------------------------------
// Kernel 3: reduce 32 tile partials -> mean
// ---------------------------------------------------------------------------
__global__ void crf_reduce_kernel(const float* __restrict__ partial,
                                  float* __restrict__ out, int n) {
  float acc = 0.f;
  for (int i = threadIdx.x; i < n; i += 32) acc += partial[i];
  for (int off = 16; off; off >>= 1) acc += __shfl_down(acc, off, 32);
  if (threadIdx.x == 0) out[0] = acc / (float)BATCH;
}

extern "C" void kernel_launch(void* const* d_in, const int* in_sizes, int n_in,
                              void* d_out, int out_size, void* d_ws, size_t ws_size,
                              hipStream_t stream) {
  const float* em   = (const float*)d_in[0];   // (512, 512, 64) f32
  const int*   tags = (const int*)d_in[1];     // (512, 512) int
  const float* mask = (const float*)d_in[2];   // (512, 512) f32
  const float* tr   = (const float*)d_in[3];   // (66, 66) f32

  float* gold    = (float*)d_ws;               // BATCH floats
  float* partial = gold + BATCH;               // BATCH/TILE_B floats

  crf_gold_kernel<<<BATCH / 8, 256, 0, stream>>>(em, tags, mask, tr, gold);
  crf_scan_kernel<<<BATCH / TILE_B, THREADS, 0, stream>>>(em, mask, tr, gold, partial);
  crf_reduce_kernel<<<1, 32, 0, stream>>>(partial, (float*)d_out, BATCH / TILE_B);
}
